// CnnSelfAttention_25331717112491
// MI455X (gfx1250) — matrix operand order repaired and checked by reading, not compiled, not run on hardware
//
#include <hip/hip_runtime.h>

typedef __attribute__((ext_vector_type(16))) _Float16 v16h;
typedef __attribute__((ext_vector_type(8)))  float    v8f;
typedef __attribute__((ext_vector_type(4)))  float    f32x4;
typedef __attribute__((ext_vector_type(4)))  unsigned int u32x4;
typedef __attribute__((ext_vector_type(8)))  int      i32x8;
typedef __attribute__((ext_vector_type(4)))  int      i32x4;

union U16h { v16h h; f32x4 f4[2]; };

#define WMMA_F16(A, B, Cv) \
  __builtin_amdgcn_wmma_f32_16x16x32_f16(false, (A), false, (B), (short)0, (Cv), false, false)

static __device__ __forceinline__ v8f vzero8() {
  v8f z;
#pragma unroll
  for (int i = 0; i < 8; ++i) z[i] = 0.f;
  return z;
}

// Build a 2-D TDM descriptor (D#) and issue TENSOR_LOAD_TO_LDS.
// Units for dims/strides: elements of data_size (2 bytes here).
// Toolchain exposes the 6-arg builtin: (g0 u32x4, g1 i32x8, g2 i32x4, g3 i32x4, i32x8, cpol)
static __device__ __forceinline__ void tdm_load_2d(
    unsigned int lds_addr, const void* gaddr,
    unsigned int tensor_d0, unsigned int tensor_d1,
    unsigned int tile_d0, unsigned int tile_d1,
    unsigned long long d0_stride)
{
  unsigned long long ga = (unsigned long long)gaddr;
  u32x4 g0;
  g0[0] = 1u;                                    // count=1 (valid user descriptor)
  g0[1] = lds_addr;                              // lds_addr @ bits 63:32
  g0[2] = (unsigned int)(ga & 0xffffffffu);      // global_addr[31:0]
  g0[3] = (unsigned int)((ga >> 32) & 0x01ffffffu) | 0x80000000u; // addr[56:32] | type=2
  i32x8 g1;
  g1[0] = (int)(1u << 16);                       // workgroup_mask=0, data_size=1 (2B)
  g1[1] = (int)((tensor_d0 & 0xffffu) << 16);    // tensor_dim0[15:0] @ bits 63:48
  g1[2] = (int)((tensor_d0 >> 16) | ((tensor_d1 & 0xffffu) << 16));
  g1[3] = (int)((tensor_d1 >> 16) | (tile_d0 << 16));  // tile_dim0 @ 127:112
  g1[4] = (int)(tile_d1 & 0xffffu);              // tile_dim1 (tile_dim2=0)
  g1[5] = (int)(unsigned int)(d0_stride & 0xffffffffu);       // dim0_stride[31:0]
  g1[6] = (int)(unsigned int)((d0_stride >> 32) & 0xffffu);   // dim0_stride[47:32]
  g1[7] = 0;                                     // dim1_stride = 0 (2-D tile)
  i32x4 z4;
  z4[0] = 0; z4[1] = 0; z4[2] = 0; z4[3] = 0;
  i32x8 z8;
#pragma unroll
  for (int i = 0; i < 8; ++i) z8[i] = 0;
  __builtin_amdgcn_tensor_load_to_lds(g0, g1, z4, z4, z8, 0);
}

static __device__ __forceinline__ unsigned int lds_off(const void* p) {
  return (unsigned int)(unsigned long long)p;    // low 32 bits of flat LDS addr = LDS offset
}

// ---------------------------------------------------------------------------
// Kernel 1: 1x1-conv projections + ReLU, f32 -> f16, attention-friendly layouts
//   Kt[b][n][0..31]  = relu(Wq . x)^T   (keys,    rows of 64B)
//   Qt[b][n][0..31]  = relu(Wk . x)^T   (queries, rows of 64B)
//   Vt[b][ch][n]     = relu(Wv . x)     (values,  contiguous along n)
// ---------------------------------------------------------------------------
__global__ __launch_bounds__(256) void proj_kernel(
    const float* __restrict__ x,  const float* __restrict__ Wq,
    const float* __restrict__ Wk, const float* __restrict__ Wv,
    _Float16* __restrict__ Kt, _Float16* __restrict__ Qt, _Float16* __restrict__ Vt)
{
  constexpr int C = 256, N = 4096;
  __shared__ alignas(16) _Float16 Xh[16][C];   // x tile, [token][channel], f16

  const int t  = threadIdx.x;
  const int b  = blockIdx.x >> 8;              // 256 token-tiles per batch
  const int n0 = (blockIdx.x & 255) << 4;

  {
    const int nl = t & 15;
    for (int c = t >> 4; c < C; c += 16)       // coalesced 64B rows
      Xh[nl][c] = (_Float16)x[((size_t)b * C + c) * N + n0 + nl];
  }
  __syncthreads();

  const int wave = t >> 5, lane = t & 31, g = lane >> 4, ln = lane & 15;
  const f32x4* xrow = (const f32x4*)(&Xh[ln][0]);

  for (int tile = wave; tile < 20; tile += 8) {
    if (tile < 4) {
      // f (Wq) -> Kt, g (Wk) -> Qt : OutT = X^T * W^T
      const int kind = (tile >> 1);                        // 0: Kt, 1: Qt
      const int chb  = (tile & 1) << 4;
      const float* W = kind ? Wk : Wq;
      v8f acc = vzero8();
      for (int cc = 0; cc < C; cc += 32) {
        U16h ua, ub;
        ua.f4[0] = xrow[(cc >> 3) + 2 * g];                // A = X^T (A-layout)
        ua.f4[1] = xrow[(cc >> 3) + 2 * g + 1];
        const float* wrow = W + (size_t)(chb + ln) * C + cc + g * 16;
#pragma unroll
        for (int i = 0; i < 16; ++i) ub.h[i] = (_Float16)wrow[i];  // B = W^T
        acc = WMMA_F16(ua.h, ub.h, acc);
      }
      _Float16* dst = kind ? Qt : Kt;
#pragma unroll
      for (int r = 0; r < 8; ++r) {
        float v = fmaxf((float)acc[r], 0.f);               // ReLU
        dst[((size_t)b * N + n0 + r + 8 * g) * 32 + chb + ln] = (_Float16)v;
      }
    } else {
      // h (Wv) -> Vt : Out = W * X
      const int chb = (tile - 4) << 4;
      v8f acc = vzero8();
      for (int cc = 0; cc < C; cc += 32) {
        U16h ua, ub;
        const float* wrow = Wv + (size_t)(chb + ln) * C + cc;
#pragma unroll
        for (int i = 0; i < 8; ++i) ua.h[i]     = (_Float16)wrow[g * 8 + i];
#pragma unroll
        for (int i = 0; i < 8; ++i) ua.h[8 + i] = (_Float16)wrow[16 + g * 8 + i];
        ub.f4[0] = xrow[(cc >> 3) + g];                    // B = X (B-layout)
        ub.f4[1] = xrow[(cc >> 3) + 2 + g];
        acc = WMMA_F16(ua.h, ub.h, acc);
      }
#pragma unroll
      for (int r = 0; r < 8; ++r) {
        float v = fmaxf((float)acc[r], 0.f);               // ReLU
        Vt[((size_t)b * C + chb + r + 8 * g) * N + n0 + ln] = (_Float16)v;
      }
    }
  }
}

// ---------------------------------------------------------------------------
// Kernel 2: fused flash attention + gamma scale + residual.
// K/V chunks staged in LDS by the Tensor Data Mover (double-buffered, wave 0
// issues, TENSORcnt + workgroup barrier hand-off). One wave = 16 queries x
// 256 channels; 8 waves per block share the staged K/V.
// ---------------------------------------------------------------------------
__global__ __launch_bounds__(256) void attn_kernel(
    const float* __restrict__ x, const float* __restrict__ gamma,
    const _Float16* __restrict__ Kt, const _Float16* __restrict__ Qt,
    const _Float16* __restrict__ Vt, float* __restrict__ out)
{
  constexpr int C = 256, N = 4096, NCHUNK = N / 32;
  __shared__ alignas(16) _Float16 Klds[2][32][32];    //  4 KB: 32 keys x 32 dims
  __shared__ alignas(16) _Float16 Vlds[2][C][32];     // 32 KB: 256 ch x 32 keys
  __shared__ alignas(16) _Float16 Plds[8][16][32];    //  8 KB: per-wave P staging

  const int t = threadIdx.x, wave = t >> 5, lane = t & 31, g = lane >> 4, ln = lane & 15;
  const int b  = blockIdx.x >> 5;                     // 32 query-groups (128 q) per batch
  const int q0 = ((blockIdx.x & 31) << 7) + (wave << 4);

  const _Float16* Kb = Kt + (size_t)b * N * 32;
  const _Float16* Vb = Vt + (size_t)b * C * N;

  // Q A-operand: 16 queries x 32 dims (persistent)
  U16h uq;
  const f32x4* qrow = (const f32x4*)(Qt + ((size_t)b * N + q0 + ln) * 32);
  uq.f4[0] = qrow[2 * g];
  uq.f4[1] = qrow[2 * g + 1];
  const v16h Q = uq.h;

  v8f acc[16];
#pragma unroll
  for (int j = 0; j < 16; ++j) acc[j] = vzero8();
  float Mr[8], Lr[8];
#pragma unroll
  for (int r = 0; r < 8; ++r) { Mr[r] = -3.0e38f; Lr[r] = 0.f; }
  const v8f z = vzero8();

  // Prologue: TDM-stage chunk 0 into buffer 0
  if (wave == 0) {
    tdm_load_2d(lds_off(&Klds[0][0][0]), Kb, (unsigned)N * 32, 1, 1024, 1, (unsigned)N * 32);
    tdm_load_2d(lds_off(&Vlds[0][0][0]), Vb, N, C, 32, C, N);
    __builtin_amdgcn_s_wait_tensorcnt(0);
  }
  __syncthreads();

  for (int ci = 0; ci < NCHUNK; ++ci) {
    const int buf = ci & 1;
    // Prefetch next chunk into the other buffer while this one is consumed
    if (wave == 0 && ci + 1 < NCHUNK) {
      const int kb1 = (ci + 1) * 32;
      tdm_load_2d(lds_off(&Klds[buf ^ 1][0][0]), Kb + (size_t)kb1 * 32,
                  (unsigned)N * 32, 1, 1024, 1, (unsigned)N * 32);
      tdm_load_2d(lds_off(&Vlds[buf ^ 1][0][0]), Vb + kb1, N, C, 32, C, N);
    }

    // Key B-operands from LDS: two 32x16 tiles
    U16h uk0, uk1;
    const f32x4* k0 = (const f32x4*)(&Klds[buf][ln][0]);
    uk0.f4[0] = k0[2 * g]; uk0.f4[1] = k0[2 * g + 1];
    const f32x4* k1 = (const f32x4*)(&Klds[buf][16 + ln][0]);
    uk1.f4[0] = k1[2 * g]; uk1.f4[1] = k1[2 * g + 1];

    v8f S0 = WMMA_F16(Q, uk0.h, z);    // S[q][key 0..15]
    v8f S1 = WMMA_F16(Q, uk1.h, z);    // S[q][key 16..31]

    // Online softmax: each 16-lane half holds 8 query rows (D-layout)
    float e0[8], e1[8], sc[8];
    int changed = 0;
#pragma unroll
    for (int r = 0; r < 8; ++r) {
      float m = fmaxf(S0[r], S1[r]);
      m = fmaxf(m, __shfl_xor(m, 1, 32));
      m = fmaxf(m, __shfl_xor(m, 2, 32));
      m = fmaxf(m, __shfl_xor(m, 4, 32));
      m = fmaxf(m, __shfl_xor(m, 8, 32));
      float Mn = fmaxf(Mr[r], m);
      sc[r] = __expf(Mr[r] - Mn);
      changed |= (Mn > Mr[r]) ? 1 : 0;
      Mr[r] = Mn;
      e0[r] = __expf(S0[r] - Mn);
      e1[r] = __expf(S1[r] - Mn);
      float s = e0[r] + e1[r];
      s += __shfl_xor(s, 1, 32);
      s += __shfl_xor(s, 2, 32);
      s += __shfl_xor(s, 4, 32);
      s += __shfl_xor(s, 8, 32);
      Lr[r] = Lr[r] * sc[r] + s;
    }
    if (__any(changed)) {              // rescale accumulators only when max moved
#pragma unroll
      for (int j = 0; j < 16; ++j)
#pragma unroll
        for (int r = 0; r < 8; ++r)
          acc[j][r] *= sc[r];
    }

    // D-layout -> A-layout relayout of P through LDS (intra-wave)
#pragma unroll
    for (int r = 0; r < 8; ++r) {
      Plds[wave][r + 8 * g][ln]      = (_Float16)e0[r];
      Plds[wave][r + 8 * g][16 + ln] = (_Float16)e1[r];
    }
    asm volatile("s_wait_dscnt 0" ::: "memory");
    U16h up;
    const f32x4* prow = (const f32x4*)(&Plds[wave][ln][0]);
    up.f4[0] = prow[2 * g];
    up.f4[1] = prow[2 * g + 1];
    const v16h P = up.h;

    // O += P (16q x 32keys) * V (32keys x 16ch), 16 channel tiles from LDS
#pragma unroll
    for (int j = 0; j < 16; ++j) {
      U16h uv;
      const f32x4* vrow = (const f32x4*)(&Vlds[buf][j * 16 + ln][0]);
      uv.f4[0] = vrow[2 * g]; uv.f4[1] = vrow[2 * g + 1];
      acc[j] = WMMA_F16(P, uv.h, acc[j]);
    }

    // Hand off: prefetch complete + all waves done with this buffer
    if (wave == 0) __builtin_amdgcn_s_wait_tensorcnt(0);
    __syncthreads();
  }

  // Epilogue: gamma / L, residual add, b128 stores (rows r are consecutive tokens)
  const float gm = gamma[0];
  float rgl[8];
#pragma unroll
  for (int r = 0; r < 8; ++r) rgl[r] = gm / Lr[r];
#pragma unroll
  for (int j = 0; j < 16; ++j) {
    const int ch = j * 16 + ln;
    const size_t base = ((size_t)b * C + ch) * N + q0 + 8 * g;
    const f32x4* xr = (const f32x4*)(x + base);
    f32x4 x0 = xr[0], x1 = xr[1];
    f32x4 o0, o1;
#pragma unroll
    for (int r = 0; r < 4; ++r) {
      o0[r] = fmaf(acc[j][r],     rgl[r],     x0[r]);
      o1[r] = fmaf(acc[j][4 + r], rgl[4 + r], x1[r]);
    }
    f32x4* orow = (f32x4*)(out + base);
    orow[0] = o0;
    orow[1] = o1;
  }
}

// ---------------------------------------------------------------------------
extern "C" void kernel_launch(void* const* d_in, const int* in_sizes, int n_in,
                              void* d_out, int out_size, void* d_ws, size_t ws_size,
                              hipStream_t stream) {
  (void)in_sizes; (void)n_in; (void)out_size; (void)ws_size;
  const float* x     = (const float*)d_in[0];
  const float* Wq    = (const float*)d_in[1];
  const float* Wk    = (const float*)d_in[2];
  const float* Wv    = (const float*)d_in[3];
  const float* gamma = (const float*)d_in[4];
  float* out = (float*)d_out;

  constexpr int B = 8, C = 256, N = 4096;
  _Float16* ws = (_Float16*)d_ws;
  _Float16* Kt = ws;                                   // B*N*32  f16 (2 MB)
  _Float16* Qt = ws + (size_t)B * N * 32;              // B*N*32  f16 (2 MB)
  _Float16* Vt = ws + (size_t)2 * B * N * 32;          // B*C*N   f16 (16 MB)

  proj_kernel<<<B * (N / 16), 256, 0, stream>>>(x, Wq, Wk, Wv, Kt, Qt, Vt);
  attn_kernel<<<B * (N / 128), 256, 0, stream>>>(x, gamma, Kt, Qt, Vt, out);
}
